// FeedForwardQuantum_65481071408164
// MI455X (gfx1250) — matrix operand order repaired and checked by reading, not compiled, not run on hardware
//
#include <hip/hip_runtime.h>
#include <hip/hip_bf16.h>

// FeedForwardQuantum for MI455X (gfx1250).
//
// Quantum layer reduces analytically to z[n,w] = cos(theta[w]) * cos(x[n,w])
// (product state; RX then RY per wire; <Z_w> = cos(theta_w) cos(x_w)).
// Remaining work: h = relu(z @ W1^T + b1)  [32768 x 2048]
//                 out = h @ W2^T + b2      [32768 x 1024]  (137 GFLOP -> WMMA f16)
//
// Workspace layout: [0, 128MB) h (f16, 32768x2048), [128MB, 132MB) W2 in f16.
// h + W2f16 = 132 MB < 192 MB L2, so GEMM re-reads stay on-die.

typedef __attribute__((ext_vector_type(16))) _Float16 v16h;
typedef __attribute__((ext_vector_type(8)))  float    v8f;

#define M_TOT   32768L
#define EMB     1024
#define FFN     2048
#define NWIRE   8

#define BM 128
#define BN 128
#define KC 32
#define SA 40   // padded LDS row stride in halves (80B) -> conflict-free frag gathers

union V16 { float4 f[2]; v16h h; };
union H8  { _Float16 hh[8]; float4 v; };

// ---------------------------------------------------------------------------
// Kernel 1: W2 f32 -> f16  (1024*2048 elements, 8 per thread)
// ---------------------------------------------------------------------------
__global__ __launch_bounds__(256) void w2_cvt_kernel(const float* __restrict__ W2,
                                                     _Float16* __restrict__ W2h) {
  long i = ((long)blockIdx.x * 256 + threadIdx.x) * 8;
  float4 a = *(const float4*)(W2 + i);
  float4 b = *(const float4*)(W2 + i + 4);
  H8 o;
  o.hh[0] = (_Float16)a.x; o.hh[1] = (_Float16)a.y;
  o.hh[2] = (_Float16)a.z; o.hh[3] = (_Float16)a.w;
  o.hh[4] = (_Float16)b.x; o.hh[5] = (_Float16)b.y;
  o.hh[6] = (_Float16)b.z; o.hh[7] = (_Float16)b.w;
  *(float4*)(W2h + i) = o.v;
}

// ---------------------------------------------------------------------------
// Kernel 2: z = cos(theta)*cos(x8); h = relu(z @ W1^T + b1) stored as f16.
// One block handles 8 rows; thread t owns columns [8t, 8t+8).
// ---------------------------------------------------------------------------
__global__ __launch_bounds__(256) void h_kernel(const float* __restrict__ x,
                                                const float* __restrict__ theta,
                                                const float* __restrict__ W1,
                                                const float* __restrict__ b1,
                                                _Float16* __restrict__ h) {
  __shared__ float zs[8 * NWIRE];
  const int t = threadIdx.x;
  const long n0 = (long)blockIdx.x * 8;

  if (t < 64) {
    int r = t >> 3, w = t & 7;
    zs[t] = cosf(theta[w]) * cosf(x[(n0 + r) * EMB + w]);
  }
  __syncthreads();

  const int k0 = t * 8;
  float w1r[64], b1r[8];
#pragma unroll
  for (int j = 0; j < 8; ++j) {
    b1r[j] = b1[k0 + j];
#pragma unroll
    for (int w = 0; w < NWIRE; ++w) w1r[j * 8 + w] = W1[(k0 + j) * NWIRE + w];
  }

#pragma unroll
  for (int r = 0; r < 8; ++r) {
    float zr[NWIRE];
#pragma unroll
    for (int w = 0; w < NWIRE; ++w) zr[w] = zs[r * NWIRE + w];
    H8 o;
#pragma unroll
    for (int j = 0; j < 8; ++j) {
      float acc = b1r[j];
#pragma unroll
      for (int w = 0; w < NWIRE; ++w) acc = fmaf(zr[w], w1r[j * 8 + w], acc);
      o.hh[j] = (_Float16)(acc > 0.f ? acc : 0.f);
    }
    *(float4*)(&h[(n0 + r) * FFN + k0]) = o.v;   // coalesced 16B store
  }
}

// ---------------------------------------------------------------------------
// Kernel 3: out = h @ W2h^T + b2 via v_wmma_f32_16x16x32_f16.
// 256 threads = 8 waves (4 along M x 2 along N), wave tile 32x64 (2x4 WMMA).
// Tiles staged with global_load_async_to_lds_b128 (ASYNCcnt), double-buffered:
// each wave issues exactly 4 async loads per tile; loads complete in order, so
// after issuing tile k+1's 4 loads, s_wait_asynccnt 4 => tile k is resident.
// ---------------------------------------------------------------------------
__global__ __launch_bounds__(256) void gemm_kernel(const _Float16* __restrict__ A,
                                                   const _Float16* __restrict__ Bm,
                                                   const float* __restrict__ b2,
                                                   float* __restrict__ out) {
  __shared__ _Float16 As[2][BM * SA];   // 2 x 10 KB
  __shared__ _Float16 Bs[2][BN * SA];   // 2 x 10 KB

  const int tid  = threadIdx.x;
  const int lane = tid & 31;
  const int wave = tid >> 5;
  const int wm   = wave & 3;     // M sub-tile: rows wm*32 ..
  const int wn   = wave >> 2;    // N sub-tile: cols wn*64 ..
  const int lr   = lane & 15;    // row/col within 16-tile
  const int lh   = lane >> 4;    // half-wave group

  const long m0 = (long)blockIdx.x * BM;
  const long n0 = (long)blockIdx.y * BN;

  // Global->LDS staging: A and B each 128 rows x 64B = 512 x 16B chunks,
  // 2 chunks per thread per matrix. chunk q: row = q>>2, 16B-piece = q&3.
  const int  ar0 = tid >> 2,          ac = tid & 3;
  const int  ar1 = (tid + 256) >> 2;                 // (q+256)&3 == q&3
  const long gA0 = (m0 + ar0) * FFN + ac * 8;
  const long gA1 = (m0 + ar1) * FFN + ac * 8;
  const long gB0 = (n0 + ar0) * FFN + ac * 8;
  const long gB1 = (n0 + ar1) * FFN + ac * 8;
  const int  lA0 = ar0 * SA + ac * 8;
  const int  lA1 = ar1 * SA + ac * 8;

  // Byte addresses for the async copies. For an LDS object the generic pointer
  // is {aperture_hi, lds_offset}, so the low 32 bits are the wave-relative LDS
  // byte address that the async-load VDST operand expects.
  const uint64_t gA0b = (uint64_t)(const void*)(A  + gA0);
  const uint64_t gA1b = (uint64_t)(const void*)(A  + gA1);
  const uint64_t gB0b = (uint64_t)(const void*)(Bm + gB0);
  const uint64_t gB1b = (uint64_t)(const void*)(Bm + gB1);
  uint32_t ldsA0[2], ldsA1[2], ldsB0[2], ldsB1[2];
#pragma unroll
  for (int p = 0; p < 2; ++p) {
    ldsA0[p] = (uint32_t)(size_t)(&As[p][lA0]);
    ldsA1[p] = (uint32_t)(size_t)(&As[p][lA1]);
    ldsB0[p] = (uint32_t)(size_t)(&Bs[p][lA0]);
    ldsB1[p] = (uint32_t)(size_t)(&Bs[p][lA1]);
  }

  auto issue_tile = [&](int p, int k0) {
    uint64_t kb = (uint64_t)k0 * 2;
    asm volatile("global_load_async_to_lds_b128 %0, %1, off"
                 :: "v"(ldsA0[p]), "v"(gA0b + kb) : "memory");
    asm volatile("global_load_async_to_lds_b128 %0, %1, off"
                 :: "v"(ldsA1[p]), "v"(gA1b + kb) : "memory");
    asm volatile("global_load_async_to_lds_b128 %0, %1, off"
                 :: "v"(ldsB0[p]), "v"(gB0b + kb) : "memory");
    asm volatile("global_load_async_to_lds_b128 %0, %1, off"
                 :: "v"(ldsB1[p]), "v"(gB1b + kb) : "memory");
  };

  v8f c[2][4] = {};

  issue_tile(0, 0);              // prologue: tile 0 in flight
  int p = 0;
  for (int k0 = 0; k0 < FFN; k0 += KC) {
    if (k0 + KC < FFN) {
      issue_tile(p ^ 1, k0 + KC);                       // next tile -> other buffer
      asm volatile("s_wait_asynccnt 0x4" ::: "memory"); // current tile resident
      if (k0 + 2 * KC < FFN)
        __builtin_prefetch(A + gA0 + k0 + 2 * KC, 0, 1); // deepen the h stream
    } else {
      asm volatile("s_wait_asynccnt 0x0" ::: "memory");
    }
    __syncthreads();             // everyone's async writes for buf p visible

    // A fragment (16x32 f16): lane holds row lr; halves j=0..7 -> K=8*lh..8*lh+7,
    // j=8..15 -> K=16+8*lh.. (ISA 7.12.2). Two 16B LDS reads.
    V16 fa[2], fb[4];
#pragma unroll
    for (int tM = 0; tM < 2; ++tM) {
      const _Float16* pa = &As[p][(wm * 32 + tM * 16 + lr) * SA + lh * 8];
      fa[tM].f[0] = *(const float4*)(pa);
      fa[tM].f[1] = *(const float4*)(pa + 16);
    }
    // B fragment (32x16 f16): lane holds col lr; j=0..7 -> K=8*lh+j, j=8..15 -> K=16+8*lh+(j-8).
#pragma unroll
    for (int tN = 0; tN < 4; ++tN) {
      const _Float16* pb = &Bs[p][(wn * 64 + tN * 16 + lr) * SA + lh * 8];
      fb[tN].f[0] = *(const float4*)(pb);
      fb[tN].f[1] = *(const float4*)(pb + 16);
    }

#pragma unroll
    for (int tM = 0; tM < 2; ++tM)
#pragma unroll
      for (int tN = 0; tN < 4; ++tN)
        c[tM][tN] = __builtin_amdgcn_wmma_f32_16x16x32_f16(
            false, fa[tM].h, false, fb[tN].h, (short)0, c[tM][tN], false, false);

    __syncthreads();             // buf p fully read before it is refilled
    p ^= 1;
  }

  // Epilogue: C/D layout -> lane (lr,lh), VGPR i holds (m = 8*lh + i, n = lr).
#pragma unroll
  for (int tN = 0; tN < 4; ++tN) {
    const long col = n0 + wn * 64 + tN * 16 + lr;
    const float bv = b2[col];
#pragma unroll
    for (int tM = 0; tM < 2; ++tM) {
      const long mrow = m0 + wm * 32 + tM * 16 + lh * 8;
#pragma unroll
      for (int i = 0; i < 8; ++i)
        out[(mrow + i) * EMB + col] = c[tM][tN][i] + bv;
    }
  }
}

// ---------------------------------------------------------------------------
extern "C" void kernel_launch(void* const* d_in, const int* in_sizes, int n_in,
                              void* d_out, int out_size, void* d_ws, size_t ws_size,
                              hipStream_t stream) {
  const float* x     = (const float*)d_in[0];
  const float* theta = (const float*)d_in[1];
  const float* W1    = (const float*)d_in[2];
  const float* b1    = (const float*)d_in[3];
  const float* W2    = (const float*)d_in[4];
  const float* b2    = (const float*)d_in[5];
  float* out = (float*)d_out;

  _Float16* hbuf = (_Float16*)d_ws;                                    // 128 MB
  _Float16* w2h  = (_Float16*)((char*)d_ws + (size_t)M_TOT * FFN * 2); // 4 MB

  w2_cvt_kernel<<<dim3((EMB * FFN) / (256 * 8)), 256, 0, stream>>>(W2, w2h);
  h_kernel<<<dim3((int)(M_TOT / 8)), 256, 0, stream>>>(x, theta, W1, b1, hbuf);
  gemm_kernel<<<dim3((int)(M_TOT / BM), EMB / BN), 256, 0, stream>>>(hbuf, w2h, b2, out);
}
